// AgglomerativePhyloGNN_69724499083982
// MI455X (gfx1250) — matrix operand order repaired and checked by reading, not compiled, not run on hardware
//
#include <hip/hip_runtime.h>
#include <hip/hip_bf16.h>

// ---------------- problem constants ----------------
#define BB 4
#define NN 32
#define LL 256
#define HH 64
#define PP 496    // C(32,2)
#define NI 31
#define NEGV (-1e9f)

// output layout (floats, flat, in return order)
#define ML_OFF   0
#define ML_SIZE  (BB*NI*PP)            // 61504
#define ANC_OFF  (ML_OFF+ML_SIZE)      // 61504
#define ANC_SIZE (BB*NI*LL*4)          // 126976
#define BR_OFF   (ANC_OFF+ANC_SIZE)    // 188480
#define BR_SIZE  (BB*NI*2)             // 248
#define LOSS_OFF (BR_OFF+BR_SIZE)      // 188728

typedef _Float16 f16;
typedef __attribute__((ext_vector_type(16))) _Float16 v16h;
typedef __attribute__((ext_vector_type(8)))  _Float16 v8h;
typedef __attribute__((ext_vector_type(8)))  float    v8f;

// ---- WMMA helpers (CDNA5 gfx1250, wave32) ----
__device__ __forceinline__ v8f wmma_f16(v16h a, v16h b, v8f c) {
  // D = A(16x32 f16) x B(32x16 f16) + C(16x16 f32)
  return __builtin_amdgcn_wmma_f32_16x16x32_f16(false, a, false, b, (short)0, c, false, false);
}

// A-tile (16x32 f16) loader from row-major f16 [rows][lda].
// ISA layout: lanes 0-15 -> rows m0..m0+15, K = {0..7, 16..23};
//             lanes 16-31 -> same rows,     K = {8..15, 24..31}.
__device__ __forceinline__ v16h load_a_tile(const f16* A, int lda, int m0, int k0, int lane) {
  int row  = m0 + (lane & 15);
  int koff = (lane >> 4) * 8;
  union { v16h v; v8h h[2]; } u;
  u.h[0] = *(const v8h*)(A + row * lda + k0 + koff);
  u.h[1] = *(const v8h*)(A + row * lda + k0 + 16 + koff);
  return u.v;
}

// B-tile from pre-packed weights: packed[((kt*4+nt)*32+lane)*16 + e]
__device__ __forceinline__ v16h load_b_tile(const f16* packed, int kt, int nt, int lane) {
  return *(const v16h*)(packed + (((kt * 4 + nt) * 32 + lane) << 4));
}

// ---------------- prep: pack weights, II/JJ, clades, init state ----------------
__global__ void prep_kernel(const int* __restrict__ order,
                            const float* __restrict__ Wm1, const float* __restrict__ Ws1,
                            const float* __restrict__ Wm2,
                            f16* Wm1p, f16* Ws1p, f16* Wm2p,
                            int* II, int* JJ, unsigned* lsets, int* active,
                            unsigned* clades, float* lossb) {
  int tid = threadIdx.x;
  // pack weights into the per-lane WMMA B-operand layout
  for (int idx = tid; idx < 20480; idx += 512) {
    const float* W; f16* Pk; int f;
    if (idx < 8192)       { W = Wm1; Pk = Wm1p; f = idx; }
    else if (idx < 16384) { W = Ws1; Pk = Ws1p; f = idx - 8192; }
    else                  { W = Wm2; Pk = Wm2p; f = idx - 16384; }
    int e = f & 15, lane = (f >> 4) & 31, nt = (f >> 9) & 3, kt = f >> 11;
    int row = kt * 32 + (lane >> 4) * 8 + (e & 7) + (e >> 3) * 16;
    int col = nt * 16 + (lane & 15);
    Pk[f] = (f16)W[row * 64 + col];
  }
  // upper-triangle pair indices in np.triu_indices order
  if (tid < PP) {
    int rem = tid, i = 0;
    while (rem >= NN - 1 - i) { rem -= NN - 1 - i; ++i; }
    II[tid] = i; JJ[tid] = i + 1 + rem;
  }
  // per-batch: leaf bitmasks, active flags, true clades, loss
  if (tid >= PP && tid < PP + BB) {
    int b = tid - PP;
    unsigned desc[2 * NN - 1];
    for (int n = 0; n < NN; ++n) {
      desc[n] = 1u << n;
      lsets[b * NN + n] = 1u << n;
      active[b * NN + n] = 1;
    }
    for (int s = 0; s < NI; ++s) {
      unsigned m = desc[order[(b * NI + s) * 2 + 0]] | desc[order[(b * NI + s) * 2 + 1]];
      desc[NN + s] = m;
      clades[b * NI + s] = m;
    }
    lossb[b] = 0.f;
  }
}

// ---------------- embedding: relu(onehot @ We + be) -> f16 pool ----------------
__global__ void embed_kernel(const float* __restrict__ leaf, const float* __restrict__ We,
                             const float* __restrict__ be, f16* __restrict__ poolh) {
  __shared__ float sWe[4 * HH];
  __shared__ float sbe[HH];
  int bn = blockIdx.x;                 // b*NN + n
  int s  = threadIdx.x;                // site
  if (s < 4 * HH) sWe[s] = We[s];
  if (s < HH)     sbe[s] = be[s];
  __syncthreads();
  const float* x = leaf + (bn * LL + s) * 4;
  float x0 = x[0], x1 = x[1], x2 = x[2], x3 = x[3];
  f16* o = poolh + ((size_t)bn * LL + s) * HH;
  for (int k = 0; k < HH; ++k) {
    float v = x0 * sWe[k] + x1 * sWe[64 + k] + x2 * sWe[128 + k] + x3 * sWe[192 + k] + sbe[k];
    o[k] = (f16)fmaxf(v, 0.f);
  }
}

// ---------------- means over sites per slot ----------------
__global__ void means_kernel(const f16* __restrict__ poolh, float* __restrict__ means) {
  int b = blockIdx.x;
  for (int idx = threadIdx.x; idx < NN * HH; idx += 256) {
    int n = idx >> 6, k = idx & 63;
    const f16* p = poolh + (((size_t)b * NN + n) * LL) * HH + k;
    float s = 0.f;
    for (int t = 0; t < LL; ++t) s += (float)p[t * HH];
    means[(b * NN + n) * HH + k] = s * (1.f / (float)LL);
  }
}

// ---------------- per-slot score projections: SA = pool@Ws1_top + bs1, SB = pool@Ws1_bot ----------------
// mode 0: slot = blockIdx.x & 31 (initial, grid B*32); mode 1: slot = selbuf si (grid B)
__global__ void project_kernel(const f16* __restrict__ poolh, const f16* __restrict__ Ws1p,
                               const float* __restrict__ bs1,
                               float* __restrict__ SA, float* __restrict__ SB,
                               const int* __restrict__ selbuf, int mode) {
  int b, slot;
  if (mode == 0) { b = blockIdx.x >> 5; slot = blockIdx.x & 31; }
  else           { b = blockIdx.x;      slot = selbuf[b * 4 + 1]; }
  int lane = threadIdx.x & 31, w = threadIdx.x >> 5;
  const f16* A = poolh + (((size_t)b * NN + slot) * LL) * HH;
  for (int i = 0; i < 16; ++i) {
    int t = w * 16 + i;
    int which = t >> 6;                 // 0 -> SA, 1 -> SB
    int rem = t & 63;
    int mt = rem >> 2, nt = rem & 3;
    v8f acc = {};
    for (int kk = 0; kk < 2; ++kk) {
      v16h a  = load_a_tile(A, HH, mt * 16, kk * 32, lane);
      v16h bt = load_b_tile(Ws1p, which * 2 + kk, nt, lane);
      acc = wmma_f16(a, bt, acc);
    }
    float* O = (which == 0 ? SA : SB) + (((size_t)b * NN + slot) * LL) * HH;
    int col = nt * 16 + (lane & 15);
    float bias = (which == 0) ? bs1[col] : 0.f;
    for (int r = 0; r < 8; ++r) {
      int row = mt * 16 + r + 8 * (lane >> 4);
      O[row * HH + col] = acc[r] + bias;
    }
  }
}

// ---------------- pair scores: one wave per pair ----------------
__global__ void score_kernel(const float* __restrict__ SA, const float* __restrict__ SB,
                             const float* __restrict__ Ws2, const float* __restrict__ bs2,
                             const int* __restrict__ II, const int* __restrict__ JJ,
                             const int* __restrict__ active,
                             float* __restrict__ scores, float* __restrict__ out_ml, int step) {
  __shared__ float sW[HH];
  int b = blockIdx.y;
  int tid = threadIdx.x;
  if (tid < HH) sW[tid] = Ws2[tid];
  __syncthreads();
  int p = blockIdx.x * 8 + (tid >> 5);
  int lane = tid & 31;
  int i = II[p], j = JJ[p];
  float score;
  if (!(active[b * NN + i] && active[b * NN + j])) {
    score = NEGV;
  } else {
    const float* Ai = SA + (((size_t)b * NN + i) * LL) * HH;
    const float* Bj = SB + (((size_t)b * NN + j) * LL) * HH;
    float acc = 0.f;
    for (int t = 0; t < 8; ++t) {
      int s = lane + t * 32;
      const float4* a4 = (const float4*)(Ai + s * HH);
      const float4* b4 = (const float4*)(Bj + s * HH);
      for (int q = 0; q < 16; ++q) {
        float4 a = a4[q], bb = b4[q];
        const float* wq = sW + q * 4;
        acc += fmaxf(a.x + bb.x, 0.f) * wq[0];
        acc += fmaxf(a.y + bb.y, 0.f) * wq[1];
        acc += fmaxf(a.z + bb.z, 0.f) * wq[2];
        acc += fmaxf(a.w + bb.w, 0.f) * wq[3];
      }
    }
    for (int off = 16; off > 0; off >>= 1) acc += __shfl_xor(acc, off, 32);
    score = acc * (1.f / (float)LL) + bs2[0];
  }
  if (lane == 0) {
    scores[b * PP + p] = score;
    out_ml[(b * NI + step) * PP + p] = score;  // scan output ml
  }
}

// ---------------- select: argmax, log-softmax clade CE loss, state update ----------------
__global__ void select_kernel(const float* __restrict__ scores,
                              const int* __restrict__ II, const int* __restrict__ JJ,
                              int* __restrict__ active, unsigned* __restrict__ lsets,
                              const unsigned* __restrict__ clades,
                              float* __restrict__ lossb, int* __restrict__ selbuf) {
  __shared__ float sval[256];
  __shared__ int   sidx[256];
  __shared__ float ssum[256];
  __shared__ float red[256];
  __shared__ int   cnt[256];
  int b = blockIdx.x, tid = threadIdx.x;
  const float* sc = scores + b * PP;
  // argmax (first-max tie break, matching jnp.argmax)
  float bv = -3.0e38f; int bi = PP;
  for (int p = tid; p < PP; p += 256) {
    float v = sc[p];
    if (v > bv) { bv = v; bi = p; }
  }
  sval[tid] = bv; sidx[tid] = bi;
  __syncthreads();
  for (int s = 128; s > 0; s >>= 1) {
    if (tid < s) {
      float v2 = sval[tid + s]; int i2 = sidx[tid + s];
      if (v2 > sval[tid] || (v2 == sval[tid] && i2 < sidx[tid])) { sval[tid] = v2; sidx[tid] = i2; }
    }
    __syncthreads();
  }
  float m = sval[0]; int sel = sidx[0];
  __syncthreads();
  // logsumexp
  float se = 0.f;
  for (int p = tid; p < PP; p += 256) se += expf(sc[p] - m);
  ssum[tid] = se;
  __syncthreads();
  for (int s = 128; s > 0; s >>= 1) { if (tid < s) ssum[tid] += ssum[tid + s]; __syncthreads(); }
  float lse = m + logf(ssum[0]);
  __syncthreads();
  // cross-entropy vs correct-pair mask
  float lsum = 0.f; int c = 0;
  for (int p = tid; p < PP; p += 256) {
    int i = II[p], j = JJ[p];
    if (active[b * NN + i] && active[b * NN + j]) {
      unsigned mg = lsets[b * NN + i] | lsets[b * NN + j];
      bool hit = false;
      for (int t = 0; t < NI; ++t) hit = hit || (mg == clades[b * NI + t]);
      if (hit) { lsum += sc[p] - lse; c += 1; }
    }
  }
  red[tid] = lsum; cnt[tid] = c;
  __syncthreads();
  for (int s = 128; s > 0; s >>= 1) {
    if (tid < s) { red[tid] += red[tid + s]; cnt[tid] += cnt[tid + s]; }
    __syncthreads();
  }
  if (tid == 0) {
    if (cnt[0] > 0) lossb[b] += -red[0] / (float)cnt[0];
    int si = II[sel], sj = JJ[sel];
    selbuf[b * 4 + 0] = sel; selbuf[b * 4 + 1] = si; selbuf[b * 4 + 2] = sj;
    lsets[b * NN + si] |= lsets[b * NN + sj];
    lsets[b * NN + sj] = 0u;
    active[b * NN + sj] = 0;
  }
}

// ---------------- merge: 2-layer parent MLP (WMMA), anc, branch lengths, state update ----------------
__global__ void merge_kernel(f16* __restrict__ poolh,
                             const f16* __restrict__ Wm1p, const f16* __restrict__ Wm2p,
                             const float* __restrict__ bm1, const float* __restrict__ bm2,
                             const float* __restrict__ Wd, const float* __restrict__ bd,
                             const float* __restrict__ Wb1, const float* __restrict__ bb1,
                             const float* __restrict__ Wb2, const float* __restrict__ bb2,
                             float* __restrict__ means, const int* __restrict__ selbuf,
                             float* __restrict__ out_anc, float* __restrict__ out_br, int step) {
  __shared__ f16 sh[LL * HH];      // 32KB: h1, then parents
  __shared__ float spm[HH];
  __shared__ float smi[HH], smj[HH];
  __shared__ float rb1[HH], rb2[HH];
  int b = blockIdx.x;
  int si = selbuf[b * 4 + 1], sj = selbuf[b * 4 + 2];
  int tid = threadIdx.x, lane = tid & 31, w = tid >> 5;
  const f16* Ai = poolh + (((size_t)b * NN + si) * LL) * HH;
  const f16* Aj = poolh + (((size_t)b * NN + sj) * LL) * HH;
  if (tid < HH) {
    smi[tid] = means[(b * NN + si) * HH + tid];   // old means (pre-merge)
    smj[tid] = means[(b * NN + sj) * HH + tid];
  }
  // layer 1: h1 = relu([pool_si, pool_sj] @ Wm1 + bm1)  -> sh
  for (int i = 0; i < 8; ++i) {
    int mt = 2 * w + (i >> 2), nt = i & 3;
    v8f acc = {};
    for (int kt = 0; kt < 4; ++kt) {
      const f16* src = (kt < 2) ? Ai : Aj;
      v16h a  = load_a_tile(src, HH, mt * 16, (kt & 1) * 32, lane);
      v16h bt = load_b_tile(Wm1p, kt, nt, lane);
      acc = wmma_f16(a, bt, acc);
    }
    int col = nt * 16 + (lane & 15);
    float bias = bm1[col];
    for (int r = 0; r < 8; ++r) {
      int row = mt * 16 + r + 8 * (lane >> 4);
      sh[row * HH + col] = (f16)fmaxf(acc[r] + bias, 0.f);
    }
  }
  __syncthreads();
  // layer 2 into registers: parents = relu(h1 @ Wm2 + bm2)
  v8f pacc[8];
  for (int i = 0; i < 8; ++i) {
    int mt = 2 * w + (i >> 2), nt = i & 3;
    v8f acc = {};
    for (int kt = 0; kt < 2; ++kt) {
      v16h a  = load_a_tile(sh, HH, mt * 16, kt * 32, lane);
      v16h bt = load_b_tile(Wm2p, kt, nt, lane);
      acc = wmma_f16(a, bt, acc);
    }
    pacc[i] = acc;
  }
  __syncthreads();
  // store parents over sh and into poolh[si]
  f16* Po = poolh + (((size_t)b * NN + si) * LL) * HH;
  for (int i = 0; i < 8; ++i) {
    int mt = 2 * w + (i >> 2), nt = i & 3;
    int col = nt * 16 + (lane & 15);
    float bias = bm2[col];
    for (int r = 0; r < 8; ++r) {
      int row = mt * 16 + r + 8 * (lane >> 4);
      f16 v = (f16)fmaxf(pacc[i][r] + bias, 0.f);
      sh[row * HH + col] = v;
      Po[row * HH + col] = v;
    }
  }
  __syncthreads();
  // anc = parents @ Wd + bd  (thread = site)
  {
    const f16* pr = sh + tid * HH;
    float a0 = bd[0], a1 = bd[1], a2 = bd[2], a3 = bd[3];
    for (int k = 0; k < HH; ++k) {
      float v = (float)pr[k];
      a0 += v * Wd[k * 4 + 0]; a1 += v * Wd[k * 4 + 1];
      a2 += v * Wd[k * 4 + 2]; a3 += v * Wd[k * 4 + 3];
    }
    float* o = out_anc + ((size_t)(b * NI + step) * LL + tid) * 4;
    o[0] = a0; o[1] = a1; o[2] = a2; o[3] = a3;
  }
  // pm = parents.mean(axis=sites)
  if (tid < HH) {
    float s = 0.f;
    for (int t = 0; t < LL; ++t) s += (float)sh[t * HH + tid];
    spm[tid] = s * (1.f / (float)LL);
  }
  __syncthreads();
  // branch lengths: softplus(relu([pm, mean] @ Wb1 + bb1) @ Wb2 + bb2)
  if (tid < HH) {
    float hi = bb1[tid], hj = bb1[tid];
    for (int c = 0; c < HH; ++c) {
      float wv = Wb1[c * HH + tid];
      hi += spm[c] * wv; hj += spm[c] * wv;
    }
    for (int c = 0; c < HH; ++c) {
      float wv = Wb1[(HH + c) * HH + tid];
      hi += smi[c] * wv; hj += smj[c] * wv;
    }
    rb1[tid] = fmaxf(hi, 0.f) * Wb2[tid];
    rb2[tid] = fmaxf(hj, 0.f) * Wb2[tid];
    means[(b * NN + si) * HH + tid] = spm[tid];   // update means after use
  }
  __syncthreads();
  if (tid == 0) {
    float s1 = 0.f, s2 = 0.f;
    for (int k = 0; k < HH; ++k) { s1 += rb1[k]; s2 += rb2[k]; }
    float x1 = s1 + bb2[0], x2 = s2 + bb2[0];
    out_br[b * (NI * 2) + step * 2 + 0] = fmaxf(x1, 0.f) + log1pf(expf(-fabsf(x1)));
    out_br[b * (NI * 2) + step * 2 + 1] = fmaxf(x2, 0.f) + log1pf(expf(-fabsf(x2)));
  }
}

__global__ void finalize_kernel(const float* __restrict__ lossb, float* __restrict__ out) {
  if (threadIdx.x == 0)
    out[0] = (lossb[0] + lossb[1] + lossb[2] + lossb[3]) * 0.25f;
}

// ---------------- host orchestration ----------------
extern "C" void kernel_launch(void* const* d_in, const int* in_sizes, int n_in,
                              void* d_out, int out_size, void* d_ws, size_t ws_size,
                              hipStream_t stream) {
  (void)in_sizes; (void)n_in; (void)out_size; (void)ws_size;
  const float* leaf  = (const float*)d_in[0];
  const int*   order = (const int*)  d_in[1];
  const float* We  = (const float*)d_in[2];
  const float* be  = (const float*)d_in[3];
  const float* Wm1 = (const float*)d_in[4];
  const float* bm1 = (const float*)d_in[5];
  const float* Wm2 = (const float*)d_in[6];
  const float* bm2 = (const float*)d_in[7];
  const float* Ws1 = (const float*)d_in[8];
  const float* bs1 = (const float*)d_in[9];
  const float* Ws2 = (const float*)d_in[10];
  const float* bs2 = (const float*)d_in[11];
  const float* Wd  = (const float*)d_in[12];
  const float* bd  = (const float*)d_in[13];
  const float* Wb1 = (const float*)d_in[14];
  const float* bb1 = (const float*)d_in[15];
  const float* Wb2 = (const float*)d_in[16];
  const float* bb2 = (const float*)d_in[17];
  float* out = (float*)d_out;
  float* out_ml  = out + ML_OFF;
  float* out_anc = out + ANC_OFF;
  float* out_br  = out + BR_OFF;
  float* out_ls  = out + LOSS_OFF;

  // workspace carve-out
  char* ws = (char*)d_ws;
  size_t off = 0;
  auto alloc = [&](size_t bytes) -> void* {
    void* p = (void*)(ws + off);
    off += (bytes + 255) & ~((size_t)255);
    return p;
  };
  f16*      poolh  = (f16*)     alloc((size_t)BB * NN * LL * HH * sizeof(f16));   // 4 MB
  float*    SA     = (float*)   alloc((size_t)BB * NN * LL * HH * sizeof(float)); // 8 MB
  float*    SB     = (float*)   alloc((size_t)BB * NN * LL * HH * sizeof(float)); // 8 MB
  f16*      Wm1p   = (f16*)     alloc(128 * 64 * sizeof(f16));
  f16*      Ws1p   = (f16*)     alloc(128 * 64 * sizeof(f16));
  f16*      Wm2p   = (f16*)     alloc(64 * 64 * sizeof(f16));
  float*    scores = (float*)   alloc(BB * PP * sizeof(float));
  int*      II     = (int*)     alloc(PP * sizeof(int));
  int*      JJ     = (int*)     alloc(PP * sizeof(int));
  unsigned* lsets  = (unsigned*)alloc(BB * NN * sizeof(unsigned));
  int*      active = (int*)     alloc(BB * NN * sizeof(int));
  unsigned* clades = (unsigned*)alloc(BB * NI * sizeof(unsigned));
  int*      selbuf = (int*)     alloc(BB * 4 * sizeof(int));
  float*    means  = (float*)   alloc(BB * NN * HH * sizeof(float));
  float*    lossb  = (float*)   alloc(BB * sizeof(float));

  prep_kernel<<<1, 512, 0, stream>>>(order, Wm1, Ws1, Wm2, Wm1p, Ws1p, Wm2p,
                                     II, JJ, lsets, active, clades, lossb);
  embed_kernel<<<BB * NN, LL, 0, stream>>>(leaf, We, be, poolh);
  means_kernel<<<BB, 256, 0, stream>>>(poolh, means);
  project_kernel<<<BB * NN, 256, 0, stream>>>(poolh, Ws1p, bs1, SA, SB, selbuf, 0);

  for (int step = 0; step < NI; ++step) {
    score_kernel<<<dim3(PP / 8, BB), 256, 0, stream>>>(SA, SB, Ws2, bs2, II, JJ, active,
                                                       scores, out_ml, step);
    select_kernel<<<BB, 256, 0, stream>>>(scores, II, JJ, active, lsets, clades, lossb, selbuf);
    merge_kernel<<<BB, 256, 0, stream>>>(poolh, Wm1p, Wm2p, bm1, bm2, Wd, bd,
                                         Wb1, bb1, Wb2, bb2, means, selbuf,
                                         out_anc, out_br, step);
    project_kernel<<<BB, 256, 0, stream>>>(poolh, Ws1p, bs1, SA, SB, selbuf, 1);
  }
  finalize_kernel<<<1, 32, 0, stream>>>(lossb, out_ls);
}
